// Seq2seq_34170759806992
// MI455X (gfx1250) — matrix operand order repaired and checked by reading, not compiled, not run on hardware
//
#include <hip/hip_runtime.h>
#include <hip/hip_bf16.h>

// ---------------------------------------------------------------------------
// Types for CDNA5 WMMA (gfx1250, wave32)
// ---------------------------------------------------------------------------
typedef __bf16 bf16_t;
typedef __bf16 v16bf __attribute__((ext_vector_type(16)));
typedef __bf16 v8bf  __attribute__((ext_vector_type(8)));
typedef float  v8f_t __attribute__((ext_vector_type(8)));

#define DEV_INLINE __device__ __forceinline__

// One K-step worth of fragments: shared B tile + 4 A tiles (64-row strip).
struct frag_set {
    v16bf b, a0, a1, a2, a3;
};

DEV_INLINE v16bf ld_pair(const bf16_t* p0, const bf16_t* p1) {
    union { v16bf v; v8bf h[2]; } u;
    u.h[0] = *(const v8bf*)p0;
    u.h[1] = *(const v8bf*)p1;
    return u.v;
}

DEV_INLINE frag_set load_frags(const bf16_t* arow0, size_t astride16,
                               const bf16_t* wrow, int k0, int hsel) {
    frag_set f;
    // B 32x16 (ISA 7.12.2): lanes 0-15 col N=lane hold K in {k0+0..7, k0+16..23},
    //                       lanes 16-31 hold K in {k0+8..15, k0+24..31}.
    f.b = ld_pair(wrow + k0 + 8 * hsel, wrow + k0 + 16 + 8 * hsel);
    // A 16x32: lanes 0-15 row M=lane hold K in {k0+0..7, k0+16..23},
    //          lanes 16-31 hold K in {k0+8..15, k0+24..31}.
    const bf16_t* ap = arow0 + k0;
    f.a0 = ld_pair(ap + 8 * hsel, ap + 16 + 8 * hsel); ap += astride16;
    f.a1 = ld_pair(ap + 8 * hsel, ap + 16 + 8 * hsel); ap += astride16;
    f.a2 = ld_pair(ap + 8 * hsel, ap + 16 + 8 * hsel); ap += astride16;
    f.a3 = ld_pair(ap + 8 * hsel, ap + 16 + 8 * hsel);
    return f;
}

// ---------------------------------------------------------------------------
// bf16 WMMA GEMM, M-blocked by 4 tiles (64 rows) per wave, double-buffered:
//   C[M x N] (f32) = A[M x K] (bf16, row-major) @ W[N x K]^T (bf16, row-major)
//                    (+ bias[N] if bias != nullptr)
// One wave32 computes a 64x16 strip: 4 x v_wmma_f32_16x16x32_bf16 per K-step
// sharing a single B fragment (each W element loaded once per GEMM). The
// fragment set for K-step i+1 is loaded before the WMMAs of step i issue, so
// loads stay in flight under the matrix ops instead of s_wait_loadcnt 0.
// ---------------------------------------------------------------------------
__global__ void gemm_bf16_wmma(const bf16_t* __restrict__ A, int lda,
                               const bf16_t* __restrict__ W, int ldw,
                               float* __restrict__ C, long long ldc,
                               const float* __restrict__ bias,
                               int Mb,   // number of 64-row M blocks
                               int Nt,   // number of 16-col N tiles
                               int K) {
    int wave = blockIdx.x * (blockDim.x >> 5) + (threadIdx.x >> 5);
    if (wave >= Mb * Nt) return;                 // wave-uniform guard
    int lane = threadIdx.x & 31;
    int bm = wave / Nt, tn = wave % Nt;
    int m0 = bm << 6, n0 = tn << 4;
    int r    = lane & 15;
    int hsel = lane >> 4;

    const bf16_t* arow0 = A + (size_t)(m0 + r) * lda;
    size_t astride16 = (size_t)16 * lda;         // 16 rows
    const bf16_t* wrow = W + (size_t)(n0 + r) * ldw;

    v8f_t acc0 = {}, acc1 = {}, acc2 = {}, acc3 = {};

    frag_set cur = load_frags(arow0, astride16, wrow, 0, hsel);
    for (int k0 = 32; k0 < K; k0 += 32) {
        // WGP-scope prefetch of the W stream one K-block ahead (W is read
        // exactly once per GEMM; pulls the next line L2 -> WGP$).
        __builtin_prefetch(wrow + k0 + 32, 0, 3);
        frag_set nxt = load_frags(arow0, astride16, wrow, k0, hsel);
        acc0 = __builtin_amdgcn_wmma_f32_16x16x32_bf16(
                   false, cur.a0, false, cur.b, (short)0, acc0, false, false);
        acc1 = __builtin_amdgcn_wmma_f32_16x16x32_bf16(
                   false, cur.a1, false, cur.b, (short)0, acc1, false, false);
        acc2 = __builtin_amdgcn_wmma_f32_16x16x32_bf16(
                   false, cur.a2, false, cur.b, (short)0, acc2, false, false);
        acc3 = __builtin_amdgcn_wmma_f32_16x16x32_bf16(
                   false, cur.a3, false, cur.b, (short)0, acc3, false, false);
        cur = nxt;
    }
    acc0 = __builtin_amdgcn_wmma_f32_16x16x32_bf16(
               false, cur.a0, false, cur.b, (short)0, acc0, false, false);
    acc1 = __builtin_amdgcn_wmma_f32_16x16x32_bf16(
               false, cur.a1, false, cur.b, (short)0, acc1, false, false);
    acc2 = __builtin_amdgcn_wmma_f32_16x16x32_bf16(
               false, cur.a2, false, cur.b, (short)0, acc2, false, false);
    acc3 = __builtin_amdgcn_wmma_f32_16x16x32_bf16(
               false, cur.a3, false, cur.b, (short)0, acc3, false, false);

    float bv = bias ? bias[n0 + r] : 0.0f;
    size_t coff = (size_t)(n0 + r);
    #pragma unroll
    for (int i = 0; i < 8; ++i) {
        int row = m0 + i + (hsel << 3);
        C[(size_t)(row)      * (size_t)ldc + coff] = acc0[i] + bv;
        C[(size_t)(row + 16) * (size_t)ldc + coff] = acc1[i] + bv;
        C[(size_t)(row + 32) * (size_t)ldc + coff] = acc2[i] + bv;
        C[(size_t)(row + 48) * (size_t)ldc + coff] = acc3[i] + bv;
    }
}

// ---------------------------------------------------------------------------
// Elementwise / support kernels
// ---------------------------------------------------------------------------
__global__ void f32_to_bf16_k(const float* __restrict__ in,
                              bf16_t* __restrict__ out, int n) {
    int i = blockIdx.x * blockDim.x + threadIdx.x;
    if (i < n) out[i] = (bf16_t)in[i];
}

__global__ void zero_f32_k(float* p, int n) {
    int i = blockIdx.x * blockDim.x + threadIdx.x;
    if (i < n) p[i] = 0.0f;
}
__global__ void zero_bf16_k(bf16_t* p, int n) {
    int i = blockIdx.x * blockDim.x + threadIdx.x;
    if (i < n) p[i] = (bf16_t)0.0f;
}
__global__ void zero_i32_k(int* p, int n) {
    int i = blockIdx.x * blockDim.x + threadIdx.x;
    if (i < n) p[i] = 0;
}

// Encoder source embedding gather, time-major rows (t*B + b), bf16 output.
__global__ void gather_src_seq_k(const int* __restrict__ x,
                                 const float* __restrict__ emb,
                                 bf16_t* __restrict__ seq,
                                 int B, int S, int H) {
    long long i = (long long)blockIdx.x * blockDim.x + threadIdx.x;
    long long total = (long long)S * B * H;
    if (i >= total) return;
    int h = (int)(i % H);
    long long row = i / H;
    int b = (int)(row % B);
    int t = (int)(row / B);
    int tok = x[b * S + t];
    seq[i] = (bf16_t)emb[(size_t)tok * H + h];
}

// Decoder token embedding gather (rows = b), bf16 output.
__global__ void gather_tok_k(const int* __restrict__ tok,
                             const float* __restrict__ emb,
                             bf16_t* __restrict__ out, int B, int H) {
    int i = blockIdx.x * blockDim.x + threadIdx.x;
    if (i >= B * H) return;
    int b = i / H, h = i % H;
    out[i] = (bf16_t)emb[(size_t)tok[b] * H + h];
}

// Fused GRU gate combine (PyTorch r,z,n ordering).
// gi/gh: (B, 3H) f32.  Writes new h (f32 + bf16), optionally:
//   seq_next : bf16 row block (next-layer input, rows = b, pre-offset by t*B*H)
//   enc_slice: f32 at enc_out + t*H, strided by S*H per batch row.
__global__ void gru_combine_k(const float* __restrict__ gi,
                              const float* __restrict__ gh,
                              const float* __restrict__ bih,
                              const float* __restrict__ bhh,
                              const float* __restrict__ hprev,
                              float* __restrict__ hout,
                              bf16_t* __restrict__ hbf,
                              bf16_t* __restrict__ seq_next,
                              float* __restrict__ enc_slice,
                              int B, int H, int S) {
    int i = blockIdx.x * blockDim.x + threadIdx.x;
    if (i >= B * H) return;
    int b = i / H, j = i % H;
    size_t g = (size_t)b * 3 * H;
    float ir  = gi[g + j]         + bih[j];
    float iz  = gi[g + H + j]     + bih[H + j];
    float in_ = gi[g + 2 * H + j] + bih[2 * H + j];
    float hr  = gh[g + j]         + bhh[j];
    float hz  = gh[g + H + j]     + bhh[H + j];
    float hn  = gh[g + 2 * H + j] + bhh[2 * H + j];
    float rg = 1.0f / (1.0f + __expf(-(ir + hr)));
    float zg = 1.0f / (1.0f + __expf(-(iz + hz)));
    float ng = tanhf(in_ + rg * hn);
    float hnew = (1.0f - zg) * ng + zg * hprev[i];
    hout[i] = hnew;
    hbf[i]  = (bf16_t)hnew;
    if (seq_next)  seq_next[i] = (bf16_t)hnew;
    if (enc_slice) enc_slice[(size_t)b * S * H + j] = hnew;
}

// v_norm[b*S+s] = max(||enc_out[b,s,:]||, eps); one block per row.
__global__ void row_norm_k(const float* __restrict__ enc_out,
                           float* __restrict__ vn, int H) {
    __shared__ float red[256];
    int row = blockIdx.x, tid = threadIdx.x;
    const float* p = enc_out + (size_t)row * H;
    float s = 0.0f;
    for (int j = tid; j < H; j += 256) { float v = p[j]; s += v * v; }
    red[tid] = s; __syncthreads();
    for (int off = 128; off; off >>= 1) {
        if (tid < off) red[tid] += red[tid + off];
        __syncthreads();
    }
    if (tid == 0) vn[row] = fmaxf(sqrtf(red[0]), 1e-8f);
}

// Cosine attention + softmax + context; also builds bf16 [out, a] concat and
// writes attention weights to the attn slice of d_out. One block per batch row.
__global__ void attention_step_k(const float* __restrict__ enc_out,
                                 const float* __restrict__ vn,
                                 const float* __restrict__ out,
                                 float* __restrict__ attn_base,
                                 bf16_t* __restrict__ concat,
                                 int t, int H, int S, int L) {
    __shared__ float red[256];
    __shared__ float w[32];
    __shared__ float qn_s;
    int b = blockIdx.x, tid = threadIdx.x;
    const float* ob = out + (size_t)b * H;

    float s = 0.0f;
    for (int j = tid; j < H; j += 256) { float v = ob[j]; s += v * v; }
    red[tid] = s; __syncthreads();
    for (int off = 128; off; off >>= 1) {
        if (tid < off) red[tid] += red[tid + off];
        __syncthreads();
    }
    if (tid == 0) qn_s = fmaxf(sqrtf(red[0]), 1e-8f);
    __syncthreads();

    for (int src = 0; src < S; ++src) {
        const float* eb = enc_out + ((size_t)b * S + src) * H;
        float d = 0.0f;
        for (int j = tid; j < H; j += 256) d += eb[j] * ob[j];
        red[tid] = d; __syncthreads();
        for (int off = 128; off; off >>= 1) {
            if (tid < off) red[tid] += red[tid + off];
            __syncthreads();
        }
        if (tid == 0) w[src] = red[0] / (vn[b * S + src] * qn_s);
        __syncthreads();
    }

    if (tid == 0) {
        float mx = -3.4e38f;
        for (int k = 0; k < S; ++k) mx = fmaxf(mx, w[k]);
        float sum = 0.0f;
        for (int k = 0; k < S; ++k) { w[k] = __expf(w[k] - mx); sum += w[k]; }
        float inv = 1.0f / sum;
        for (int k = 0; k < S; ++k) {
            w[k] *= inv;
            attn_base[((size_t)b * L + t) * S + k] = w[k];
        }
    }
    __syncthreads();

    for (int j = tid; j < H; j += 256) {
        float a = 0.0f;
        for (int k = 0; k < S; ++k)
            a += w[k] * enc_out[((size_t)b * S + k) * H + j];
        concat[(size_t)b * 2 * H + j]     = (bf16_t)ob[j];
        concat[(size_t)b * 2 * H + H + j] = (bf16_t)a;
    }
}

// Greedy argmax over V logits at logits_base + b*ldc; first-max tie rule.
__global__ void argmax_row_k(const float* __restrict__ logits_base,
                             long long ldc, int V, int* __restrict__ tok) {
    __shared__ float bv[256];
    __shared__ int   bi[256];
    int b = blockIdx.x, tid = threadIdx.x;
    const float* p = logits_base + (size_t)b * (size_t)ldc;
    float best = -3.4e38f; int bidx = 0;
    for (int v = tid; v < V; v += 256) {
        float xv = p[v];
        if (xv > best) { best = xv; bidx = v; }
    }
    bv[tid] = best; bi[tid] = bidx; __syncthreads();
    for (int off = 128; off; off >>= 1) {
        if (tid < off) {
            if (bv[tid + off] > bv[tid] ||
                (bv[tid + off] == bv[tid] && bi[tid + off] < bi[tid])) {
                bv[tid] = bv[tid + off]; bi[tid] = bi[tid + off];
            }
        }
        __syncthreads();
    }
    if (tid == 0) tok[b] = bi[0];
}

// ---------------------------------------------------------------------------
// Host-side orchestration
// ---------------------------------------------------------------------------
extern "C" void kernel_launch(void* const* d_in, const int* in_sizes, int n_in,
                              void* d_out, int out_size, void* d_ws, size_t ws_size,
                              hipStream_t stream) {
    (void)in_sizes; (void)n_in; (void)out_size; (void)ws_size;
    const int H = 1024, NL = 2, B = 64, S = 32, L = 32, V = 32000;
    const int H3 = 3 * H;

    const int*   x        = (const int*)  d_in[0];
    const float* fr_emb   = (const float*)d_in[1];
    const float* en_emb   = (const float*)d_in[2];
    const float* enc_Wih  = (const float*)d_in[3];
    const float* enc_Whh  = (const float*)d_in[4];
    const float* enc_bih  = (const float*)d_in[5];
    const float* enc_bhh  = (const float*)d_in[6];
    const float* dec_Wih  = (const float*)d_in[7];
    const float* dec_Whh  = (const float*)d_in[8];
    const float* dec_bih  = (const float*)d_in[9];
    const float* dec_bhh  = (const float*)d_in[10];
    const float* fcW      = (const float*)d_in[11];
    const float* fcb      = (const float*)d_in[12];

    float* out_f   = (float*)d_out;
    float* vec_out = out_f;                                  // (B, L, V)
    float* hfin    = out_f + (size_t)B * L * V;              // (NL, B, H)
    float* attn    = hfin + (size_t)NL * B * H;              // (B, L, S)

    // ---- workspace carve-out (256B aligned) ----
    char* wsp = (char*)d_ws;
    auto carve = [&](size_t bytes) -> char* {
        char* p = wsp;
        wsp += (bytes + 255) & ~(size_t)255;
        return p;
    };
    bf16_t* encWih_bf = (bf16_t*)carve((size_t)NL * H3 * H * 2);
    bf16_t* encWhh_bf = (bf16_t*)carve((size_t)NL * H3 * H * 2);
    bf16_t* decWih_bf = (bf16_t*)carve((size_t)NL * H3 * H * 2);
    bf16_t* decWhh_bf = (bf16_t*)carve((size_t)NL * H3 * H * 2);
    bf16_t* fcW_bf    = (bf16_t*)carve((size_t)V * 2 * H * 2);   // 131 MB: fits L2
    bf16_t* seqA      = (bf16_t*)carve((size_t)S * B * H * 2);
    bf16_t* seqB      = (bf16_t*)carve((size_t)S * B * H * 2);
    float*  gi_all    = (float*) carve((size_t)S * B * H3 * 4);  // also decoder gi
    float*  gh_buf    = (float*) carve((size_t)B * H3 * 4);
    float*  enc_out   = (float*) carve((size_t)B * S * H * 4);
    float*  h0        = (float*) carve((size_t)B * H * 4);
    float*  h1        = (float*) carve((size_t)B * H * 4);
    bf16_t* h0bf      = (bf16_t*)carve((size_t)B * H * 2);
    bf16_t* h1bf      = (bf16_t*)carve((size_t)B * H * 2);
    bf16_t* emb_bf    = (bf16_t*)carve((size_t)B * H * 2);
    bf16_t* concat_bf = (bf16_t*)carve((size_t)B * 2 * H * 2);
    float*  vnorm     = (float*) carve((size_t)B * S * 4);
    int*    tok       = (int*)   carve((size_t)B * 4);

    float*  hbuf[2]  = { h0, h1 };
    bf16_t* hbfb[2]  = { h0bf, h1bf };

    auto cvt = [&](const float* src, bf16_t* dst, long long n) {
        int blocks = (int)((n + 255) / 256);
        f32_to_bf16_k<<<blocks, 256, 0, stream>>>(src, dst, (int)n);
    };
    // M must be a multiple of 64 (true for all calls here: 64 or 2048).
    auto gemm = [&](const bf16_t* A, int lda, const bf16_t* W, int ldw,
                    float* C, long long ldc, const float* bias,
                    int M, int Nt, int K) {
        int Mb = M / 64;
        int waves = Mb * Nt;
        int blocks = (waves + 7) / 8;     // 8 wave32 per 256-thread block
        gemm_bf16_wmma<<<blocks, 256, 0, stream>>>(A, lda, W, ldw, C, ldc,
                                                   bias, Mb, Nt, K);
    };

    // ---- 1) convert weights to bf16 (once per call, deterministic) ----
    cvt(enc_Wih, encWih_bf, (long long)NL * H3 * H);
    cvt(enc_Whh, encWhh_bf, (long long)NL * H3 * H);
    cvt(dec_Wih, decWih_bf, (long long)NL * H3 * H);
    cvt(dec_Whh, decWhh_bf, (long long)NL * H3 * H);
    cvt(fcW,     fcW_bf,    (long long)V * 2 * H);

    // ---- 2) encoder ----
    {
        long long n = (long long)S * B * H;
        gather_src_seq_k<<<(int)((n + 255) / 256), 256, 0, stream>>>(
            x, fr_emb, seqA, B, S, H);
    }
    bf16_t* seq_cur = seqA;
    bf16_t* seq_nxt = seqB;
    for (int l = 0; l < NL; ++l) {
        zero_f32_k <<<(B * H + 255) / 256, 256, 0, stream>>>(hbuf[l], B * H);
        zero_bf16_k<<<(B * H + 255) / 256, 256, 0, stream>>>(hbfb[l], B * H);
        // batched input-gate GEMM over all timesteps: (S*B, 3H)
        gemm(seq_cur, H, encWih_bf + (size_t)l * H3 * H, H,
             gi_all, H3, nullptr, S * B, H3 / 16, H);
        for (int t = 0; t < S; ++t) {
            gemm(hbfb[l], H, encWhh_bf + (size_t)l * H3 * H, H,
                 gh_buf, H3, nullptr, B, H3 / 16, H);
            gru_combine_k<<<(B * H + 255) / 256, 256, 0, stream>>>(
                gi_all + (size_t)t * B * H3, gh_buf,
                enc_bih + (size_t)l * H3, enc_bhh + (size_t)l * H3,
                hbuf[l], hbuf[l], hbfb[l],
                (l < NL - 1) ? seq_nxt + (size_t)t * B * H : nullptr,
                (l == NL - 1) ? enc_out + (size_t)t * H : nullptr,
                B, H, S);
        }
        bf16_t* tmp = seq_cur; seq_cur = seq_nxt; seq_nxt = tmp;
    }
    row_norm_k<<<B * S, 256, 0, stream>>>(enc_out, vnorm, H);

    // ---- 3) greedy decoder with cosine attention ----
    zero_i32_k<<<1, B, 0, stream>>>(tok, B);
    for (int t = 0; t < L; ++t) {
        gather_tok_k<<<(B * H + 255) / 256, 256, 0, stream>>>(
            tok, en_emb, emb_bf, B, H);
        const bf16_t* xin = emb_bf;
        for (int l = 0; l < NL; ++l) {
            gemm(xin, H, decWih_bf + (size_t)l * H3 * H, H,
                 gi_all, H3, nullptr, B, H3 / 16, H);
            gemm(hbfb[l], H, decWhh_bf + (size_t)l * H3 * H, H,
                 gh_buf, H3, nullptr, B, H3 / 16, H);
            gru_combine_k<<<(B * H + 255) / 256, 256, 0, stream>>>(
                gi_all, gh_buf,
                dec_bih + (size_t)l * H3, dec_bhh + (size_t)l * H3,
                hbuf[l], hbuf[l], hbfb[l], nullptr, nullptr, B, H, S);
            xin = hbfb[l];
        }
        attention_step_k<<<B, 256, 0, stream>>>(
            enc_out, vnorm, hbuf[NL - 1], attn, concat_bf, t, H, S, L);
        // FC head: logits straight into vec_out[b, t, :] (ldc = L*V)
        gemm(concat_bf, 2 * H, fcW_bf, 2 * H,
             vec_out + (size_t)t * V, (long long)L * V, fcb,
             B, V / 16, 2 * H);
        argmax_row_k<<<B, 256, 0, stream>>>(
            vec_out + (size_t)t * V, (long long)L * V, V, tok);
    }

    // ---- 4) final hidden state ----
    hipMemcpyAsync(hfin,                 h0, (size_t)B * H * 4,
                   hipMemcpyDeviceToDevice, stream);
    hipMemcpyAsync(hfin + (size_t)B * H, h1, (size_t)B * H * 4,
                   hipMemcpyDeviceToDevice, stream);
}
